// AttentionBlock_1477468750190
// MI455X (gfx1250) — matrix operand order repaired and checked by reading, not compiled, not run on hardware
//
#include <hip/hip_runtime.h>
#include <stdint.h>

#define S_LEN  4096
#define HIDDEN 256
#define NHEADS 8
#define DHEAD  32

typedef __attribute__((ext_vector_type(16))) __bf16   v16bf;
typedef __attribute__((ext_vector_type(8)))  __bf16   v8bf;
typedef __attribute__((ext_vector_type(8)))  float    v8f;
typedef __attribute__((ext_vector_type(4)))  uint32_t u32x4;
typedef __attribute__((ext_vector_type(8)))  int      i32x8;
typedef __attribute__((ext_vector_type(4)))  int      i32x4;

// ---------------------------------------------------------------------------
// TDM: issue a 1-D/2-D tensor_load_to_lds.  Descriptor layout per CDNA5 ISA
// ch.8: group0 = {count, lds_addr, global_addr, type=2}; group1 = {data_size,
// tensor dims, tile dims, dim0 stride}.  data_size = 8 bytes/element.
// tile_d1 == 0 -> 1-D copy.  stride0 in 8-byte units.
// This toolchain exposes the 6-arg builtin: (g0, g1, g2, g3, g4, cpol).
// ---------------------------------------------------------------------------
__device__ __forceinline__ void tdm_load_to_lds(uint32_t lds_off, const void* gaddr,
                                                uint32_t tile_d0, uint32_t tile_d1,
                                                uint32_t tensor_d0, uint32_t tensor_d1,
                                                uint32_t stride0) {
    uint64_t ga = (uint64_t)gaddr;
    u32x4 g0;
    g0[0] = 1u;                                            // count=1, user descriptor
    g0[1] = lds_off;                                       // lds_addr (bytes)
    g0[2] = (uint32_t)ga;                                  // global_addr[31:0]
    g0[3] = (uint32_t)((ga >> 32) & 0x01FFFFFFu) | (2u << 30); // addr[56:32] | type=2
    i32x8 g1;
    g1[0] = (int)(3u << 16);                               // data_size=3 (8B), wg_mask=0
    g1[1] = (int)((tensor_d0 & 0xFFFFu) << 16);            // tensor_dim0[15:0]
    g1[2] = (int)((tensor_d0 >> 16) | ((tensor_d1 & 0xFFFFu) << 16));
    g1[3] = (int)((tensor_d1 >> 16) | (tile_d0 << 16));    // tile_dim0
    g1[4] = (int)(tile_d1 & 0xFFFFu);                      // tile_dim1 (0 -> 1-D)
    g1[5] = (int)stride0;                                  // tensor_dim0_stride[31:0]
    g1[6] = 0;
    g1[7] = 0;
    i32x4 zz4 = {0, 0, 0, 0};
    i32x8 zz8 = {0, 0, 0, 0, 0, 0, 0, 0};
    __builtin_amdgcn_tensor_load_to_lds(g0, g1, zz4, zz4, zz8, 0);
}

// ---------------------------------------------------------------------------
// Kernel 0: convert 4 weight matrices [256x256] fp32 row-major (k,n) into
// bf16 transposed Wt[n][k] so WMMA B-fragments are contiguous per-lane loads.
// ---------------------------------------------------------------------------
__global__ __launch_bounds__(256)
void prep_weights_kernel(const float* __restrict__ Wq, const float* __restrict__ Wk,
                         const float* __restrict__ Wv, const float* __restrict__ Wo,
                         __bf16* __restrict__ Wt) {
    int idx = blockIdx.x * blockDim.x + threadIdx.x;   // 0 .. 4*65536
    int w  = idx >> 16;
    int e  = idx & 65535;
    int nn = e >> 8;
    int kk = e & 255;
    const float* W = (w == 0) ? Wq : (w == 1) ? Wk : (w == 2) ? Wv : Wo;
    Wt[(size_t)w * 65536 + e] = (__bf16)W[kk * HIDDEN + nn];
}

// ---------------------------------------------------------------------------
// Kernel 1: projection GEMM  out = X[8192,256] @ W + b.  One wave = one
// 16x64 output strip: A fragment loaded once per K-step, 4 WMMAs.
// VTRANS=0 -> write bf16 [B,H,S,D]; VTRANS=1 -> write bf16 [B,H,D,S].
// ---------------------------------------------------------------------------
template <int VTRANS>
__global__ __launch_bounds__(128)
void proj_kernel(const float* __restrict__ X, const __bf16* __restrict__ Wt,
                 const float* __restrict__ bias, __bf16* __restrict__ out) {
    const int lane = threadIdx.x & 31;
    const int wv   = threadIdx.x >> 5;
    const int n = lane & 15;          // A: row M / B,C: col N
    const int g = lane >> 4;          // lane half
    const int r0 = (blockIdx.x * 4 + wv) * 16;  // row tile over B*S
    const int n0 = blockIdx.y * 64;   // col strip over HID

    v8f acc[4] = {};
    for (int k0 = 0; k0 < HIDDEN; k0 += 32) {
        // A fragment: lane holds row n, K chunk [16g,16g+16)
        v16bf a;
        const float* xr = X + (size_t)(r0 + n) * HIDDEN + k0 + g * 16;
#pragma unroll
        for (int i = 0; i < 8; i++) { a[i] = (__bf16)xr[i]; a[8 + i] = (__bf16)xr[8 + i]; }
#pragma unroll
        for (int t = 0; t < 4; t++) {
            v16bf b = *(const v16bf*)(Wt + (size_t)(n0 + 16 * t + n) * HIDDEN + k0 + g * 16);
            acc[t] = __builtin_amdgcn_wmma_f32_16x16x32_bf16(false, a, false, b,
                                                             (short)0, acc[t], false, false);
        }
    }

#pragma unroll
    for (int t = 0; t < 4; t++) {
        const int col = n0 + 16 * t + n;
        const float bv = bias[col];
        const int hh = col >> 5;      // head
        const int d  = col & 31;      // dim in head
#pragma unroll
        for (int r = 0; r < 8; r++) {
            int row = r0 + r + 8 * g; // flat over B*S
            int bb = row >> 12;       // / 4096
            int si = row & 4095;
            float val = acc[t][r] + bv;
            size_t idx;
            if (VTRANS) idx = (((size_t)bb * NHEADS + hh) * DHEAD + d) * S_LEN + si;
            else        idx = (((size_t)bb * NHEADS + hh) * S_LEN + si) * DHEAD + d;
            out[idx] = (__bf16)val;
        }
    }
}

// ---------------------------------------------------------------------------
// Kernel 2: flash attention.  4 waves/block share one (b,h); each wave owns a
// 16-query block.  Per 64-key chunk: wave 0 DMAs the K chunk (contiguous 4KB)
// and V chunk (32x128B strided) into LDS via the Tensor Data Mover; all waves
// then run 4 score WMMAs, online softmax, P relayout via LDS, 4 PV WMMAs.
// ---------------------------------------------------------------------------
__global__ __launch_bounds__(128)
void attn_kernel(const __bf16* __restrict__ Q, const __bf16* __restrict__ K,
                 const __bf16* __restrict__ Vt, const int* __restrict__ mask,
                 __bf16* __restrict__ Wout) {
    __shared__ __align__(16) __bf16 plds[4][16][64];   // per-wave P relayout tiles
    __shared__ __align__(16) __bf16 ldsK[64][DHEAD];   // K chunk: 64 keys x 32 d
    __shared__ __align__(16) __bf16 ldsV[DHEAD][64];   // V chunk: 32 d x 64 keys

    const int lane = threadIdx.x & 31;
    const int wv   = threadIdx.x >> 5;
    const int n = lane & 15;
    const int g = lane >> 4;
    const int b = blockIdx.z;
    const int h = blockIdx.y;
    const int sq0 = (blockIdx.x * 4 + wv) * 16;

    const __bf16* Qp = Q  + ((size_t)b * NHEADS + h) * S_LEN * DHEAD;
    const __bf16* Kp = K  + ((size_t)b * NHEADS + h) * S_LEN * DHEAD;
    const __bf16* Vp = Vt + ((size_t)b * NHEADS + h) * DHEAD * S_LEN;
    const int*    mp = mask + (size_t)b * S_LEN;

    const uint32_t ldsK_off = (uint32_t)(uintptr_t)(void*)&ldsK[0][0];
    const uint32_t ldsV_off = (uint32_t)(uintptr_t)(void*)&ldsV[0][0];

    // Q A-fragment (fixed for whole loop): lane row = n, K chunk 16g..16g+15
    v16bf qa;
    {
        const __bf16* row = Qp + (size_t)(sq0 + n) * DHEAD + g * 16;
        v8bf lo = *(const v8bf*)(row);
        v8bf hi = *(const v8bf*)(row + 8);
#pragma unroll
        for (int i = 0; i < 8; i++) { qa[i] = lo[i]; qa[8 + i] = hi[i]; }
    }

    v8f acc0 = {}, acc1 = {};
    float mrow[8], lrow[8];
#pragma unroll
    for (int r = 0; r < 8; r++) { mrow[r] = -1e30f; lrow[r] = 0.0f; }

    const float rscale = 0.17677669529663687f;   // 1/sqrt(32)

    for (int sk0 = 0; sk0 < S_LEN; sk0 += 64) {
        // --- stage K/V chunk into LDS via TDM (wave 0 only) ---------------
        __syncthreads();                       // prior chunk fully consumed
        if (wv == 0) {
            // K: 64 consecutive rows of [S,32] bf16 = 4KB contiguous -> 1-D
            tdm_load_to_lds(ldsK_off, Kp + (size_t)sk0 * DHEAD,
                            512, 0, 512, 0, 512);
            // V: 32 rows x 128B, row stride S_LEN*2 bytes -> 2-D tile
            tdm_load_to_lds(ldsV_off, Vp + sk0,
                            16, 32, 16, 32, S_LEN / 4);
        }
        __builtin_amdgcn_s_wait_tensorcnt(0);
        __syncthreads();                       // chunk visible to all waves

        // --- scores: 4 x 16-wide key tiles from LDS -----------------------
        v16bf kb[4];
        int mk[4];
#pragma unroll
        for (int t = 0; t < 4; t++) {
            kb[t] = *(const v16bf*)(&ldsK[16 * t + n][g * 16]);
            mk[t] = mp[sk0 + 16 * t + n];
        }
        v8f z = {};
        v8f s[4];
#pragma unroll
        for (int t = 0; t < 4; t++)
            s[t] = __builtin_amdgcn_wmma_f32_16x16x32_bf16(false, qa, false, kb[t],
                                                           (short)0, z, false, false);

        // --- online softmax, P -> LDS as bf16 ----------------------------
#pragma unroll
        for (int r = 0; r < 8; r++) {
            float v[4];
#pragma unroll
            for (int t = 0; t < 4; t++) v[t] = mk[t] ? s[t][r] * rscale : -1e30f;
            float tm = fmaxf(fmaxf(v[0], v[1]), fmaxf(v[2], v[3]));
#pragma unroll
            for (int xm = 1; xm < 16; xm <<= 1) tm = fmaxf(tm, __shfl_xor(tm, xm, 32));
            float mnew  = fmaxf(mrow[r], tm);
            float alpha = __expf(mrow[r] - mnew);
            float rs = 0.0f;
#pragma unroll
            for (int t = 0; t < 4; t++) {
                float e = mk[t] ? __expf(v[t] - mnew) : 0.0f;
                rs += e;
                plds[wv][r + 8 * g][16 * t + n] = (__bf16)e;
            }
#pragma unroll
            for (int xm = 1; xm < 16; xm <<= 1) rs += __shfl_xor(rs, xm, 32);
            lrow[r] = lrow[r] * alpha + rs;
            mrow[r] = mnew;
            acc0[r] *= alpha;
            acc1[r] *= alpha;
        }

        // --- C-fragment -> A-fragment relayout through LDS ----------------
        asm volatile("s_wait_dscnt 0" ::: "memory");
        v16bf pa0, pa1;
        {
            const __bf16* prow = &plds[wv][n][g * 16];
            v8bf lo0 = *(const v8bf*)(prow);
            v8bf hi0 = *(const v8bf*)(prow + 8);
            v8bf lo1 = *(const v8bf*)(prow + 32);
            v8bf hi1 = *(const v8bf*)(prow + 40);
#pragma unroll
            for (int i = 0; i < 8; i++) {
                pa0[i] = lo0[i]; pa0[8 + i] = hi0[i];
                pa1[i] = lo1[i]; pa1[8 + i] = hi1[i];
            }
        }

        // --- P @ V: V B-fragments from LDS V tile ------------------------
        v16bf vb00 = *(const v16bf*)(&ldsV[n]     [     g * 16]);
        v16bf vb01 = *(const v16bf*)(&ldsV[16 + n][     g * 16]);
        v16bf vb10 = *(const v16bf*)(&ldsV[n]     [32 + g * 16]);
        v16bf vb11 = *(const v16bf*)(&ldsV[16 + n][32 + g * 16]);
        acc0 = __builtin_amdgcn_wmma_f32_16x16x32_bf16(false, pa0, false, vb00,
                                                       (short)0, acc0, false, false);
        acc0 = __builtin_amdgcn_wmma_f32_16x16x32_bf16(false, pa1, false, vb10,
                                                       (short)0, acc0, false, false);
        acc1 = __builtin_amdgcn_wmma_f32_16x16x32_bf16(false, pa0, false, vb01,
                                                       (short)0, acc1, false, false);
        acc1 = __builtin_amdgcn_wmma_f32_16x16x32_bf16(false, pa1, false, vb11,
                                                       (short)0, acc1, false, false);
    }

    // finalize: divide by row sums, store weighted as bf16 [B*S, 256]
#pragma unroll
    for (int r = 0; r < 8; r++) {
        int srow = sq0 + r + 8 * g;
        float inv = 1.0f / lrow[r];
        size_t base = ((size_t)b * S_LEN + srow) * HIDDEN + h * DHEAD;
        Wout[base + n]      = (__bf16)(acc0[r] * inv);
        Wout[base + 16 + n] = (__bf16)(acc1[r] * inv);
    }
}

// ---------------------------------------------------------------------------
// Kernel 3: output projection  out = weighted[8192,256] @ Wo + bo  (fp32 out)
// One wave = 16x64 output strip, A fragment reused across 4 WMMAs per K-step.
// ---------------------------------------------------------------------------
__global__ __launch_bounds__(128)
void out_proj_kernel(const __bf16* __restrict__ Xbf, const __bf16* __restrict__ Wt,
                     const float* __restrict__ bias, float* __restrict__ out) {
    const int lane = threadIdx.x & 31;
    const int wv   = threadIdx.x >> 5;
    const int n = lane & 15;
    const int g = lane >> 4;
    const int r0 = (blockIdx.x * 4 + wv) * 16;
    const int n0 = blockIdx.y * 64;

    v8f acc[4] = {};
    for (int k0 = 0; k0 < HIDDEN; k0 += 32) {
        v16bf a;
        const __bf16* xr = Xbf + (size_t)(r0 + n) * HIDDEN + k0 + g * 16;
        v8bf lo = *(const v8bf*)(xr);
        v8bf hi = *(const v8bf*)(xr + 8);
#pragma unroll
        for (int i = 0; i < 8; i++) { a[i] = lo[i]; a[8 + i] = hi[i]; }
#pragma unroll
        for (int t = 0; t < 4; t++) {
            v16bf b = *(const v16bf*)(Wt + (size_t)(n0 + 16 * t + n) * HIDDEN + k0 + g * 16);
            acc[t] = __builtin_amdgcn_wmma_f32_16x16x32_bf16(false, a, false, b,
                                                             (short)0, acc[t], false, false);
        }
    }
#pragma unroll
    for (int t = 0; t < 4; t++) {
        const float bv = bias[n0 + 16 * t + n];
#pragma unroll
        for (int r = 0; r < 8; r++)
            out[(size_t)(r0 + r + 8 * g) * HIDDEN + n0 + 16 * t + n] = acc[t][r] + bv;
    }
}

// ---------------------------------------------------------------------------
extern "C" void kernel_launch(void* const* d_in, const int* in_sizes, int n_in,
                              void* d_out, int out_size, void* d_ws, size_t ws_size,
                              hipStream_t stream) {
    const float* query = (const float*)d_in[0];
    const float* key   = (const float*)d_in[1];
    const float* value = (const float*)d_in[2];
    const int*   mask  = (const int*)d_in[3];
    const float* Wq = (const float*)d_in[4];
    const float* bq = (const float*)d_in[5];
    const float* Wk = (const float*)d_in[6];
    const float* bk = (const float*)d_in[7];
    const float* Wv = (const float*)d_in[8];
    const float* bv = (const float*)d_in[9];
    const float* Wo = (const float*)d_in[10];
    const float* bo = (const float*)d_in[11];

    const size_t ROWS = (size_t)2 * S_LEN;       // 8192
    char* ws = (char*)d_ws;
    __bf16* Wt   = (__bf16*)ws;                  // 4 * 65536 bf16 = 512 KB
    __bf16* Qbf  = (__bf16*)(ws + (512 << 10));  // 4 MB each below
    __bf16* Kbf  = Qbf  + ROWS * HIDDEN;
    __bf16* Vtbf = Kbf  + ROWS * HIDDEN;
    __bf16* Wgt  = Vtbf + ROWS * HIDDEN;

    prep_weights_kernel<<<(4 * 65536) / 256, 256, 0, stream>>>(Wq, Wk, Wv, Wo, Wt);

    dim3 gproj(ROWS / 64, HIDDEN / 64);          // 128 x 4 blocks, 4 waves each
    proj_kernel<0><<<gproj, 128, 0, stream>>>(query, Wt + 0 * 65536, bq, Qbf);
    proj_kernel<0><<<gproj, 128, 0, stream>>>(key,   Wt + 1 * 65536, bk, Kbf);
    proj_kernel<1><<<gproj, 128, 0, stream>>>(value, Wt + 2 * 65536, bv, Vtbf);

    dim3 gattn(S_LEN / 16 / 4, NHEADS, 2);       // 64 x 8 x 2, 4 waves/block
    attn_kernel<<<gattn, 128, 0, stream>>>(Qbf, Kbf, Vtbf, mask, Wgt);

    out_proj_kernel<<<gproj, 128, 0, stream>>>(Wgt, Wt + 3 * 65536, bo, (float*)d_out);
}
